// GAT_9259949490771
// MI455X (gfx1250) — matrix operand order repaired and checked by reading, not compiled, not run on hardware
//
#include <hip/hip_runtime.h>
#include <hip/hip_bf16.h>
#include <math.h>
#include <stdint.h>

// ---------------------------------------------------------------------------
// Types for CDNA5 WMMA (wave32, 16x16x32 f16 -> f32 accumulate)
// ---------------------------------------------------------------------------
typedef __attribute__((ext_vector_type(16))) _Float16 v16h;
typedef __attribute__((ext_vector_type(8)))  _Float16 v8h;
typedef __attribute__((ext_vector_type(8)))  float    v8f;

#define D_FEAT 128

// ---------------------------------------------------------------------------
// f32 -> f16 conversion (hoisted out of the GEMM hot loop)
// ---------------------------------------------------------------------------
__global__ void gat_f32_to_f16(const float* __restrict__ in,
                               _Float16* __restrict__ out, int n) {
    const int i = blockIdx.x * blockDim.x + threadIdx.x;
    if (i < n) out[i] = (_Float16)in[i];
}

// ---------------------------------------------------------------------------
// Pack W[128,128] (row-major f32) into WMMA B-fragment order, f16.
// B-fragment layout (16-bit B, 32x16 per k-step, ISA 7.12.2):
//   lane l: N = l&15, e<8 -> K = kbase + 8*(l>>4) + e, e>=8 -> K = kbase + 16 + 8*(l>>4) + (e-8).
// Linear layout: Wf[((ct*4 + kt)*32 + lane)*16 + e], e = K offset 0..15.
// One thread per (ct,kt,lane) = 8*4*32 = 1024 threads.
// ---------------------------------------------------------------------------
__global__ void gat_pack_wfrag(const float* __restrict__ W,
                               _Float16* __restrict__ Wf) {
    const int t = blockIdx.x * blockDim.x + threadIdx.x;
    if (t >= 1024) return;
    const int lane = t & 31;
    const int kt   = (t >> 5) & 3;
    const int ct   = t >> 7;
    const int n     = ct * 16 + (lane & 15);
    const int kbase = kt * 32 + (lane >> 4) * 8;
    _Float16 frag[16];
#pragma unroll
    for (int e = 0; e < 16; ++e)
        frag[e] = (_Float16)W[(size_t)(kbase + e + (e >= 8 ? 8 : 0)) * D_FEAT + n];
    v8h* dst = (v8h*)(Wf + (size_t)t * 16);
    dst[0] = *(const v8h*)(frag);
    dst[1] = *(const v8h*)(frag + 8);
}

// ---------------------------------------------------------------------------
// GEMM: H[N,128] = Xh[N,128] @ W (pre-packed fragments), f16 WMMA, f32 acc.
// Block = 256 threads = 8 waves; wave w computes column tile w.
// A tile (16x128 f16, 4KB) staged straight into LDS with the CDNA5 async
// global->LDS path (ASYNCcnt-tracked, no VGPR round trip), then read back as
// ds_load_b128 fragments. B fragments are two coalesced global_load_b128 per
// k-step from the packed buffer (32KB, L2-resident).
// N must be a multiple of 16 (50000 is).
// ---------------------------------------------------------------------------
__global__ void gat_gemm_wmma(const _Float16* __restrict__ Xh,
                              const _Float16* __restrict__ Wf,
                              float* __restrict__ H) {
    __shared__ _Float16 sX[16 * D_FEAT];   // 4KB A tile
    const int tid = threadIdx.x;
    const int m0  = blockIdx.x * 16;

    // Cooperative async stage: 256 lanes x 16B = 4KB directly into LDS.
    // VDST = LDS byte address (generic->LDS truncates to addr[31:0] per the
    // aperture rules), VADDR = 64-bit global address.
    {
        const _Float16* gsrc = Xh + (size_t)m0 * D_FEAT + tid * 8;
        const unsigned   ldsoff = (unsigned)(uintptr_t)(sX + tid * 8);
        asm volatile("global_load_async_to_lds_b128 %0, %1, off"
                     :: "v"(ldsoff), "v"(gsrc) : "memory");
        asm volatile("s_wait_asynccnt 0x0" ::: "memory");
    }
    __syncthreads();

    const int wave = tid >> 5;
    const int lane = tid & 31;
    const int half = lane >> 4;     // lane group 0/1
    const int lr   = lane & 15;     // A: row-in-tile, B/D: column-in-tile

    v8f acc = {};
#pragma unroll
    for (int kt = 0; kt < 4; ++kt) {
        // A-fragment (16-bit A 16x32, ISA 7.12.2):
        //   lane half h: K = 32kt + 16*(v>=4) + 8h + 2*(v&3) + {0,1}
        //   -> halves [0..7]  = K 32kt+8h    .. +7  (contiguous)
        //   -> halves [8..15] = K 32kt+8h+16 .. +7  (contiguous)
        const _Float16* ar = sX + lr * D_FEAT + kt * 32 + half * 8;
        const v8h alo = *(const v8h*)ar;
        const v8h ahi = *(const v8h*)(ar + 16);
        const v16h a = __builtin_shufflevector(alo, ahi,
            0, 1, 2, 3, 4, 5, 6, 7, 8, 9, 10, 11, 12, 13, 14, 15);

        // B-fragment: 32B contiguous per lane from the packed buffer
        const v8h* bp = (const v8h*)(Wf + ((size_t)(wave * 4 + kt) * 32 + lane) * 16);
        const v8h blo = bp[0];
        const v8h bhi = bp[1];
        const v16h b = __builtin_shufflevector(blo, bhi,
            0, 1, 2, 3, 4, 5, 6, 7, 8, 9, 10, 11, 12, 13, 14, 15);

        acc = __builtin_amdgcn_wmma_f32_16x16x32_f16(
            /*neg_a=*/false, a, /*neg_b=*/false, b,
            /*c_mod=*/(short)0, acc, /*reuse_a=*/false, /*reuse_b=*/false);
    }

    // C/D layout: VGPR r -> M = r + 8*half, N = lr
#pragma unroll
    for (int r = 0; r < 8; ++r) {
        H[(size_t)(m0 + r + half * 8) * D_FEAT + wave * 16 + lr] = acc[r];
    }
}

// ---------------------------------------------------------------------------
// Per-node attention scores: as[i] = h[i].a_src, ad[i] = h[i].a_dst
// ---------------------------------------------------------------------------
__global__ void gat_node_scores(const float* __restrict__ H,
                                const float* __restrict__ a_src,
                                const float* __restrict__ a_dst,
                                float* __restrict__ as_out,
                                float* __restrict__ ad_out, int N) {
    const int i = blockIdx.x * blockDim.x + threadIdx.x;
    if (i >= N) return;
    const float4* h = (const float4*)(H + (size_t)i * D_FEAT);
    float s = 0.f, d = 0.f;
#pragma unroll 4
    for (int k = 0; k < D_FEAT / 4; ++k) {
        const float4 hv = h[k];
        const float4 sv = ((const float4*)a_src)[k];
        const float4 dv = ((const float4*)a_dst)[k];
        s += hv.x * sv.x + hv.y * sv.y + hv.z * sv.z + hv.w * sv.w;
        d += hv.x * dv.x + hv.y * dv.y + hv.z * dv.z + hv.w * dv.w;
    }
    as_out[i] = s;
    ad_out[i] = d;
}

// ---------------------------------------------------------------------------
// Init kernels (must run every call: atomics accumulate into these buffers)
// ---------------------------------------------------------------------------
__global__ void gat_zero_f32(float* __restrict__ p, int n) {
    const int i = blockIdx.x * blockDim.x + threadIdx.x;
    if (i < n) p[i] = 0.f;
}

__global__ void gat_init_ms(float* __restrict__ m, float* __restrict__ s, int N) {
    const int i = blockIdx.x * blockDim.x + threadIdx.x;
    if (i < N) { m[i] = -INFINITY; s[i] = 0.f; }
}

// ---------------------------------------------------------------------------
// Edge pass 1: e = leaky_relu(as[src] + ad[dst]); segment-max into m[dst]
// (float atomic max via signed/unsigned int compare trick; m init'd to -inf)
// ---------------------------------------------------------------------------
__global__ void gat_edge_max(const long long* __restrict__ src,
                             const long long* __restrict__ dst,
                             const float* __restrict__ as_in,
                             const float* __restrict__ ad_in,
                             float* __restrict__ e_out,
                             float* __restrict__ m, int E) {
    const int i = blockIdx.x * blockDim.x + threadIdx.x;
    if (i >= E) return;
    const int s = (int)src[i];
    const int d = (int)dst[i];
    float v = as_in[s] + ad_in[d];
    v = (v > 0.f) ? v : 0.2f * v;          // leaky_relu, negative_slope = 0.2
    e_out[i] = v;
    if (v >= 0.f) atomicMax((int*)(m + d), __float_as_int(v));
    else          atomicMin((unsigned int*)(m + d), __float_as_uint(v));
}

// ---------------------------------------------------------------------------
// Edge pass 2: ex = exp(e - m[dst]); segment-sum into s[dst]
// ---------------------------------------------------------------------------
__global__ void gat_edge_exp(const long long* __restrict__ dst,
                             const float* __restrict__ e_in,
                             const float* __restrict__ m,
                             float* __restrict__ ex_out,
                             float* __restrict__ ssum, int E) {
    const int i = blockIdx.x * blockDim.x + threadIdx.x;
    if (i >= E) return;
    const int d = (int)dst[i];
    const float x = __expf(e_in[i] - m[d]);
    ex_out[i] = x;
    atomicAdd(ssum + d, x);
}

// ---------------------------------------------------------------------------
// Edge pass 3 (heavy, L2-resident): out[dst] += (ex/s[dst]) * h[src]
// One wave per edge; lane handles 4 consecutive features via float4 gather.
// ---------------------------------------------------------------------------
__global__ void gat_edge_aggregate(const long long* __restrict__ src,
                                   const long long* __restrict__ dst,
                                   const float* __restrict__ ex_in,
                                   const float* __restrict__ ssum,
                                   const float* __restrict__ H,
                                   float* __restrict__ out, int E) {
    const int gid  = blockIdx.x * blockDim.x + threadIdx.x;
    const int e    = gid >> 5;
    const int lane = gid & 31;
    if (e >= E) return;
    const int s = (int)src[e];
    const int d = (int)dst[e];
    const float alpha = ex_in[e] / (ssum[d] + 1e-16f);
    const float4 hv = ((const float4*)(H + (size_t)s * D_FEAT))[lane];
    float* o = out + (size_t)d * D_FEAT + lane * 4;
    atomicAdd(o + 0, alpha * hv.x);
    atomicAdd(o + 1, alpha * hv.y);
    atomicAdd(o + 2, alpha * hv.z);
    atomicAdd(o + 3, alpha * hv.w);
}

// ---------------------------------------------------------------------------
// Layer-1 epilogue: t = relu(0.1 * x + 0.9 * (agg + b1)), emitted as f16
// (GEMM input precision), conversion hoisted here instead of the GEMM loop.
// ---------------------------------------------------------------------------
__global__ void gat_epilogue1(const float* __restrict__ x,
                              const float* __restrict__ agg,
                              const float* __restrict__ b,
                              _Float16* __restrict__ th, int total) {
    const int i = blockIdx.x * blockDim.x + threadIdx.x;
    if (i >= total) return;
    const float v = 0.1f * x[i] + 0.9f * (agg[i] + b[i & (D_FEAT - 1)]);
    th[i] = (_Float16)((v > 0.f) ? v : 0.f);
}

// ---------------------------------------------------------------------------
// Layer-2 epilogue: out += b2
// ---------------------------------------------------------------------------
__global__ void gat_add_bias(float* __restrict__ out,
                             const float* __restrict__ b, int total) {
    const int i = blockIdx.x * blockDim.x + threadIdx.x;
    if (i < total) out[i] += b[i & (D_FEAT - 1)];
}

// ---------------------------------------------------------------------------
// Orchestration
// ---------------------------------------------------------------------------
extern "C" void kernel_launch(void* const* d_in, const int* in_sizes, int n_in,
                              void* d_out, int out_size, void* d_ws, size_t ws_size,
                              hipStream_t stream) {
    const float*     x   = (const float*)d_in[0];       // [N,128]
    const long long* adj = (const long long*)d_in[1];   // [2,E] int64
    // d_in[2] = edge_attr, ignored (edge_dim=None in reference)
    const float* W1  = (const float*)d_in[3];
    const float* aS1 = (const float*)d_in[4];
    const float* aD1 = (const float*)d_in[5];
    const float* b1  = (const float*)d_in[6];
    const float* W2  = (const float*)d_in[7];
    const float* aS2 = (const float*)d_in[8];
    const float* aD2 = (const float*)d_in[9];
    const float* b2  = (const float*)d_in[10];

    const int N = in_sizes[0] / D_FEAT;   // 50000 (multiple of 16)
    const int E = in_sizes[1] / 2;        // 800000
    const long long* src = adj;
    const long long* dst = adj + E;
    const int ND = N * D_FEAT;

    // Workspace layout (all offsets stay 16B-aligned for the given sizes)
    char* ws = (char*)d_ws;
    size_t off = 0;
    float* h   = (float*)(ws + off); off += (size_t)ND * sizeof(float);
    float* agg = (float*)(ws + off); off += (size_t)ND * sizeof(float);
    _Float16* xh  = (_Float16*)(ws + off); off += (size_t)ND * sizeof(_Float16);
    _Float16* th  = (_Float16*)(ws + off); off += (size_t)ND * sizeof(_Float16);
    _Float16* Wf1 = (_Float16*)(ws + off); off += (size_t)D_FEAT * D_FEAT * sizeof(_Float16);
    _Float16* Wf2 = (_Float16*)(ws + off); off += (size_t)D_FEAT * D_FEAT * sizeof(_Float16);
    float* as_b = (float*)(ws + off); off += (size_t)N * sizeof(float);
    float* ad_b = (float*)(ws + off); off += (size_t)N * sizeof(float);
    float* m_b  = (float*)(ws + off); off += (size_t)N * sizeof(float);
    float* s_b  = (float*)(ws + off); off += (size_t)N * sizeof(float);
    float* e_b  = (float*)(ws + off); off += (size_t)E * sizeof(float);
    float* ex_b = (float*)(ws + off); off += (size_t)E * sizeof(float);

    const int BT = 256;
    const dim3 blk(BT);
    const dim3 gN((N + BT - 1) / BT);
    const dim3 gE((E + BT - 1) / BT);
    const dim3 gE32(((size_t)E * 32 + BT - 1) / BT);
    const dim3 gND(((size_t)ND + BT - 1) / BT);
    const dim3 gGemm(N / 16);   // 3125 blocks, 8 waves each -> 8 col tiles
    const dim3 gPack(1024 / BT);

    // ---------------- Pre-pass: precision cast + fragment packing ----------
    gat_f32_to_f16<<<gND, blk, 0, stream>>>(x, xh, ND);
    gat_pack_wfrag<<<gPack, blk, 0, stream>>>(W1, Wf1);
    gat_pack_wfrag<<<gPack, blk, 0, stream>>>(W2, Wf2);

    // ---------------- Layer 1 ----------------
    gat_gemm_wmma<<<gGemm, blk, 0, stream>>>(xh, Wf1, h);
    gat_node_scores<<<gN, blk, 0, stream>>>(h, aS1, aD1, as_b, ad_b, N);
    gat_zero_f32<<<gND, blk, 0, stream>>>(agg, ND);
    gat_init_ms<<<gN, blk, 0, stream>>>(m_b, s_b, N);
    gat_edge_max<<<gE, blk, 0, stream>>>(src, dst, as_b, ad_b, e_b, m_b, E);
    gat_edge_exp<<<gE, blk, 0, stream>>>(dst, e_b, m_b, ex_b, s_b, E);
    gat_edge_aggregate<<<gE32, blk, 0, stream>>>(src, dst, ex_b, s_b, h, agg, E);
    gat_epilogue1<<<gND, blk, 0, stream>>>(x, agg, b1, th, ND);

    // ---------------- Layer 2 ----------------
    gat_gemm_wmma<<<gGemm, blk, 0, stream>>>(th, Wf2, h);
    gat_node_scores<<<gN, blk, 0, stream>>>(h, aS2, aD2, as_b, ad_b, N);
    gat_zero_f32<<<gND, blk, 0, stream>>>((float*)d_out, ND);
    gat_init_ms<<<gN, blk, 0, stream>>>(m_b, s_b, N);
    gat_edge_max<<<gE, blk, 0, stream>>>(src, dst, as_b, ad_b, e_b, m_b, E);
    gat_edge_exp<<<gE, blk, 0, stream>>>(dst, e_b, m_b, ex_b, s_b, E);
    gat_edge_aggregate<<<gE32, blk, 0, stream>>>(src, dst, ex_b, s_b, h, (float*)d_out, E);
    gat_add_bias<<<gND, blk, 0, stream>>>((float*)d_out, b2, ND);
}
